// EYOLOTiny_65017214927523
// MI455X (gfx1250) — compile-verified, matrix operand order checked by reading, not run on hardware
//
#include <hip/hip_runtime.h>
#include <hip/hip_bf16.h>
#include <stdint.h>

// ---------------------------------------------------------------------------
// EYOLO-tiny post-process for MI455X (gfx1250, wave32).
//   K1 decode   : per-anchor box decode + softmax-max score. The 80-class
//                 sum-of-exp is done on the matrix pipe via
//                 v_wmma_f32_16x16x32_f16 (16 anchors/wave, B = ones).
//   K2 sort     : full descending bitonic sort of 32768 u64 keys inside one
//                 WGP's LDS (256KB of the 320KB), loaded with
//                 global_load_async_to_lds_b64 (ASYNCcnt path).
//   K3 gather   : top-4096 gather -> outputs + NMS staging.
//   K4 mask     : 4096x4096 suppression bitmask (64 u64 words per column).
//   K5 nms      : sequential greedy scan in ONE wave32; keep state lives in
//                 registers (2 words/lane), __ballot for the reduction.
// ---------------------------------------------------------------------------

#define N_ANCH 25200
#define NPAD   32768
#define NTOP   4096

typedef __attribute__((ext_vector_type(16))) _Float16 v16h;
typedef __attribute__((ext_vector_type(8)))  float    v8f;

// ------------------------------ K1: decode ---------------------------------
__global__ __launch_bounds__(256) void decode_kernel(
    const float* __restrict__ ps, const float* __restrict__ pm,
    const float* __restrict__ pl, const float* __restrict__ anch,
    unsigned long long* __restrict__ keys, float* __restrict__ boxes,
    int* __restrict__ clsOut)
{
    __shared__ _Float16 expLds[8][16][96];   // [wave][anchor][class(pad to 96)]
    __shared__ float    sums[8][16];

    const int lane = threadIdx.x & 31;
    const int wave = threadIdx.x >> 5;
    const int m    = lane >> 1;        // anchor slot within wave: 0..15
    const int hv   = lane & 1;         // which half of the 80 classes
    const int gid  = blockIdx.x * 128 + wave * 16 + m;   // 0..32767

    // zero my half of the exp row (covers the 80..95 zero padding too)
    for (int t = 0; t < 48; ++t) expLds[wave][m][hv * 48 + t] = (_Float16)0.0f;

    const bool inRange = gid < N_ANCH;

    const float* pred = ps; int H = 80, W = 80, lvl = 0, within = gid;
    float stride = 8.f;
    if (gid >= 24000)      { pred = pl; H = W = 20; stride = 32.f; lvl = 2; within = gid - 24000; }
    else if (gid >= 19200) { pred = pm; H = W = 40; stride = 16.f; lvl = 1; within = gid - 19200; }
    const int a    = within % 3;
    const int cell = within / 3;
    const int x    = cell % W, y = cell / W;
    const int hw   = H * W;
    const int spat = y * W + x;

    // per-pair max + argmax over 80 class logits (each lane scans 40)
    float maxv = -3.4e38f; int amax = 0;
    if (inRange) {
        for (int c = hv * 40; c < hv * 40 + 40; ++c) {
            float v = pred[(3 + 80 * a + c) * hw + spat];
            if (v > maxv) { maxv = v; amax = c; }
        }
    }
    float om = __shfl_xor(maxv, 1);
    int   oa = __shfl_xor(amax, 1);
    if (om > maxv || (om == maxv && oa < amax)) { maxv = om; amax = oa; }

    // exp(cls - max) into LDS (f16, A-matrix staging)
    if (inRange) {
        for (int c = hv * 40; c < hv * 40 + 40; ++c) {
            float v = pred[(3 + 80 * a + c) * hw + spat];
            expLds[wave][m][c] = (_Float16)__expf(v - maxv);
        }
    }
    __syncthreads();

    // ---- sum-of-exp on the matrix pipe: D = A(16x32 exps) * B(32x16 ones) ----
    v16h ones;
    for (int e = 0; e < 16; ++e) ones[e] = (_Float16)1.0f;
    v8f acc = {};
    const _Float16* base = &expLds[wave][0][0];
    const int M   = (lane < 16) ? lane : lane - 16;
    const int kb0 = (lane < 16) ? 0 : 8;
    for (int q = 0; q < 3; ++q) {            // K chunks: 0..31, 32..63, 64..95
        v16h av;
        const _Float16* row = base + M * 96 + q * 32;
        for (int e = 0; e < 8; ++e) av[e]     = row[kb0 + e];        // K = kb0..kb0+7
        for (int e = 0; e < 8; ++e) av[8 + e] = row[16 + kb0 + e];   // K = 16+kb0..
        acc = __builtin_amdgcn_wmma_f32_16x16x32_f16(
                  false, av, false, ones, (short)0, acc, false, false);
    }
    // D layout: lanes 0..15 hold M=vgpr, lanes 16..31 hold M=vgpr+8
    if (lane == 0)  for (int e = 0; e < 8; ++e) sums[wave][e]     = acc[e];
    if (lane == 16) for (int e = 0; e < 8; ++e) sums[wave][8 + e] = acc[e];
    __syncthreads();

    if (hv == 0) {
        if (inRange) {
            float sumexp = sums[wave][m];                    // includes exp(0)=1 term
            float obj    = pred[a * hw + spat];
            float sobj   = 1.f / (1.f + __expf(-obj));
            float sc     = sobj / sumexp;                    // sigmoid(obj)*softmax_max

            float r0 = pred[(243 + 4 * a + 0) * hw + spat];
            float r1 = pred[(243 + 4 * a + 1) * hw + spat];
            float r2 = pred[(243 + 4 * a + 2) * hw + spat];
            float r3 = pred[(243 + 4 * a + 3) * hw + spat];
            float cx = (1.f / (1.f + __expf(-r0)) + (float)x) * stride;
            float cy = (1.f / (1.f + __expf(-r1)) + (float)y) * stride;
            float aw = anch[lvl * 6 + a * 2 + 0];
            float ah = anch[lvl * 6 + a * 2 + 1];
            float bw = __expf(r2) * aw;
            float bh = __expf(r3) * ah;
            const float inv = 1.f / 640.f;
            float b0 = fminf(fmaxf((cx - 0.5f * bw) * inv, 0.f), 1.f);
            float b1 = fminf(fmaxf((cy - 0.5f * bh) * inv, 0.f), 1.f);
            float b2 = fminf(fmaxf((cx + 0.5f * bw) * inv, 0.f), 1.f);
            float b3 = fminf(fmaxf((cy + 0.5f * bh) * inv, 0.f), 1.f);

            float scm = (sc >= 0.001f) ? sc : -1.0f;
            unsigned bits = __float_as_uint(scm);
            unsigned sortable = (bits & 0x80000000u) ? ~bits : (bits | 0x80000000u);
            keys[gid] = ((unsigned long long)sortable << 32) | (unsigned)(~gid);
            boxes[gid * 4 + 0] = b0; boxes[gid * 4 + 1] = b1;
            boxes[gid * 4 + 2] = b2; boxes[gid * 4 + 3] = b3;
            clsOut[gid] = amax;
        } else {
            keys[gid] = 0ull;                                // pad: below everything
        }
    }
}

// --------------------- K2: 32768-key bitonic sort in LDS -------------------
__global__ __launch_bounds__(1024) void sort_kernel(
    const unsigned long long* __restrict__ keys,
    unsigned long long* __restrict__ sorted)
{
    extern __shared__ unsigned long long sdata[];            // 256 KB dynamic LDS
    const int t = threadIdx.x;

    // async global -> LDS fill (gfx1250 ASYNCcnt path)
    unsigned ldsBase = (unsigned)(uintptr_t)(&sdata[0]);
    for (int i = 0; i < 32; ++i) {
        int idx = t + (i << 10);
        unsigned lds = ldsBase + (unsigned)idx * 8u;
        unsigned long long gp = (unsigned long long)(uintptr_t)(keys + idx);
        asm volatile("global_load_async_to_lds_b64 %0, %1, off"
                     :: "v"(lds), "v"(gp) : "memory");
    }
    asm volatile("s_wait_asynccnt 0" ::: "memory");
    __syncthreads();

    // full descending bitonic sort (15 stages, 120 passes)
    for (int k = 2; k <= NPAD; k <<= 1) {
        for (int j = k >> 1; j > 0; j >>= 1) {
            for (int p = 0; p < 32; ++p) {
                int i   = t + (p << 10);
                int ixj = i ^ j;
                if (ixj > i) {
                    unsigned long long A = sdata[i], B = sdata[ixj];
                    bool up = (i & k) == 0;
                    if ((up && A < B) || (!up && A > B)) { sdata[i] = B; sdata[ixj] = A; }
                }
            }
            __syncthreads();
        }
    }
    for (int i = t; i < NTOP; i += 1024) sorted[i] = sdata[i];
}

// ----------------------------- K3: gather ----------------------------------
__global__ __launch_bounds__(256) void gather_kernel(
    const unsigned long long* __restrict__ sorted,
    const float* __restrict__ boxes, const int* __restrict__ cls,
    float* __restrict__ out, float4* __restrict__ topBox,
    int* __restrict__ topCls, int* __restrict__ topValid)
{
    int i = blockIdx.x * 256 + threadIdx.x;
    if (i >= NTOP) return;
    unsigned long long key = sorted[i];
    unsigned s    = (unsigned)(key >> 32);
    unsigned bits = (s & 0x80000000u) ? (s & 0x7FFFFFFFu) : ~s;
    float sc = __uint_as_float(bits);
    unsigned idx = ~((unsigned)key);
    if (idx >= N_ANCH) idx = 0;
    float4 bb = ((const float4*)boxes)[idx];
    int    cc = cls[idx];
    out[i * 4 + 0] = bb.x; out[i * 4 + 1] = bb.y;
    out[i * 4 + 2] = bb.z; out[i * 4 + 3] = bb.w;
    out[4 * NTOP + i] = sc;
    out[5 * NTOP + i] = (float)cc;
    topBox[i]   = bb;
    topCls[i]   = cc;
    topValid[i] = (sc >= 0.001f) ? 1 : 0;
}

// ------------------- K4: suppression bitmask (4096 x 64 u64) ---------------
__global__ __launch_bounds__(256) void mask_kernel(
    const float4* __restrict__ topBox, const int* __restrict__ topCls,
    unsigned long long* __restrict__ mask)
{
    int tid = blockIdx.x * 256 + threadIdx.x;   // 0 .. 262143
    int j = tid >> 6;                           // column (suppressee)
    int w = tid & 63;                           // which 64-candidate word
    float4 bj = topBox[j];
    int    cj = topCls[j];
    float areaj = (bj.z - bj.x) * (bj.w - bj.y);
    unsigned long long acc = 0;
    int kbase = w << 6;
    for (int u = 0; u < 64; ++u) {
        int k = kbase + u;
        float4 bk = topBox[k];
        float xx1 = fmaxf(bj.x, bk.x), yy1 = fmaxf(bj.y, bk.y);
        float xx2 = fminf(bj.z, bk.z), yy2 = fminf(bj.w, bk.w);
        float ww = fmaxf(1e-28f, xx2 - xx1);
        float hh = fmaxf(1e-28f, yy2 - yy1);
        float inter = ww * hh;
        float areak = (bk.z - bk.x) * (bk.w - bk.y);
        float iou   = inter / (areak + areaj - inter + 1e-14f);
        bool s = (topCls[k] == cj) && (iou > 0.6f) && (k < j);
        acc |= ((unsigned long long)s) << u;
    }
    mask[j * 64 + w] = acc;
}

// -------------------- K5: sequential greedy scan, one wave32 ---------------
__global__ __launch_bounds__(32) void nms_kernel(
    const unsigned long long* __restrict__ mask,
    const int* __restrict__ topValid, float* __restrict__ out)
{
    int lane = threadIdx.x;                    // 0..31, two keep words each
    unsigned long long k0 = 0, k1 = 0;
    unsigned long long m0 = mask[2 * lane], m1 = mask[2 * lane + 1];
    for (int i = 0; i < NTOP; ++i) {
        unsigned long long n0 = 0, n1 = 0;
        if (i + 1 < NTOP) {                    // software pipeline next row
            n0 = mask[(i + 1) * 64 + 2 * lane];
            n1 = mask[(i + 1) * 64 + 2 * lane + 1];
        }
        if (i + 2 < NTOP)                      // global_prefetch two ahead
            __builtin_prefetch(&mask[(i + 2) * 64 + 2 * lane], 0, 0);
        int sup    = (((k0 & m0) | (k1 & m1)) != 0ull) ? 1 : 0;
        int anySup = (__ballot(sup) != 0ull) ? 1 : 0;
        int keep   = (topValid[i] && !anySup) ? 1 : 0;
        if (lane == (i >> 7)) {
            unsigned long long bit = ((unsigned long long)keep) << (i & 63);
            if (((i >> 6) & 1) == 0) k0 |= bit; else k1 |= bit;
        }
        if (lane == 0) out[6 * NTOP + i] = keep ? 1.0f : 0.0f;
        m0 = n0; m1 = n1;
    }
}

// ------------------------------- launcher ----------------------------------
extern "C" void kernel_launch(void* const* d_in, const int* in_sizes, int n_in,
                              void* d_out, int out_size, void* d_ws, size_t ws_size,
                              hipStream_t stream) {
    (void)in_sizes; (void)n_in; (void)out_size; (void)ws_size;
    const float* ps   = (const float*)d_in[0];
    const float* pm   = (const float*)d_in[1];
    const float* pl   = (const float*)d_in[2];
    const float* anch = (const float*)d_in[3];
    char* ws = (char*)d_ws;
    unsigned long long* keys   = (unsigned long long*)(ws + 0);        // 256 KB
    float*              boxes  = (float*)             (ws + 262144);   // 394 KB
    int*                cls    = (int*)               (ws + 665600);   // 99 KB
    unsigned long long* sorted = (unsigned long long*)(ws + 786432);   // 32 KB
    float4*             topBox = (float4*)            (ws + 851968);   // 64 KB
    int*                topCls = (int*)               (ws + 917504);   // 16 KB
    int*                topVal = (int*)               (ws + 933888);   // 16 KB
    unsigned long long* supM   = (unsigned long long*)(ws + 1048576);  // 2 MB
    float* out = (float*)d_out;

    decode_kernel<<<256, 256, 0, stream>>>(ps, pm, pl, anch, keys, boxes, cls);
    sort_kernel  <<<1, 1024, NPAD * sizeof(unsigned long long), stream>>>(keys, sorted);
    gather_kernel<<<16, 256, 0, stream>>>(sorted, boxes, cls, out, topBox, topCls, topVal);
    mask_kernel  <<<1024, 256, 0, stream>>>(topBox, topCls, supM);
    nms_kernel   <<<1, 32, 0, stream>>>(supM, topVal, out);
}
